// IGMSA_7009386627529
// MI455X (gfx1250) — matrix operand order repaired and checked by reading, not compiled, not run on hardware
//
#include <hip/hip_runtime.h>
#include <hip/hip_bf16.h>
#include <math.h>

typedef __attribute__((ext_vector_type(16))) __bf16 v16bf;
typedef __attribute__((ext_vector_type(8)))  __bf16 v8bf;
typedef __attribute__((ext_vector_type(8)))  float  v8f;

union V16 { v16bf v; struct { v8bf lo, hi; } h; };

#define NTOK 65536

static __device__ inline v8bf cvt8(const float* p) {
  v8bf r;
#pragma unroll
  for (int i = 0; i < 8; i++) r[i] = (__bf16)p[i];
  return r;
}
static __device__ inline v8bf ld8(const __bf16* p) { return *(const v8bf*)p; }

static __device__ inline v8f wmma_bf16(v16bf a, v16bf b, v8f c) {
  // D = A(16x32 bf16) x B(32x16 bf16) + C(16x16 f32)
  return __builtin_amdgcn_wmma_f32_16x16x32_bf16(false, a, false, b, (short)0, c,
                                                 false, false);
}

// ---------------- K0: zero accumulators ----------------
__global__ void k_zero(float* p, int n) {
  int i = blockIdx.x * 256 + threadIdx.x;
  if (i < n) p[i] = 0.f;
}

// ---------------- K1: QKV projection, illum gate, transpose-store, sumsq ----
// grid: 1024 blocks (2 batches x 512), block 256 (8 waves, 1 token-tile each)
__global__ __launch_bounds__(256) void k_qkv(
    const float* __restrict__ x, const float* __restrict__ illu,
    const float* __restrict__ Wq, const float* __restrict__ Wk,
    const float* __restrict__ Wv, __bf16* __restrict__ qT,
    __bf16* __restrict__ kT, __bf16* __restrict__ vT,
    float* __restrict__ vinpT, float* __restrict__ ssq_q,
    float* __restrict__ ssq_k) {
  __shared__ __attribute__((aligned(32))) __bf16 wl[4][8][32][16];  // 32 KB

  const int b = blockIdx.x >> 9;
  const int tbase = (blockIdx.x & 511) * 8;
  const int wave = threadIdx.x >> 5, lane = threadIdx.x & 31;
  const int g = lane >> 4, lr = lane & 15;
  const int token0 = (tbase + wave) * 16;
  const int row = token0 + lr;

  // A operands: 16 tokens x 128 channels, bf16, shared across q/k/v
  const float* xrow = x + ((size_t)b * NTOK + row) * 128;
  V16 a[4];
#pragma unroll
  for (int kt = 0; kt < 4; kt++) {
    const int k0 = kt * 32 + g * 16;
    a[kt].h.lo = cvt8(xrow + k0);
    a[kt].h.hi = cvt8(xrow + k0 + 8);
  }

  for (int m = 0; m < 3; m++) {
    const float* W = (m == 0) ? Wq : ((m == 1) ? Wk : Wv);
    __syncthreads();
    // pack W into WMMA-B operand order: tile(kt,nt), lane, slot s -> K=kt*32+g*16+s
#pragma unroll
    for (int i = 0; i < 4; i++) {
      const int p = threadIdx.x + 256 * i;
      const int pl = p & 31, nt = (p >> 5) & 7, kt = p >> 8;
      const int pg = pl >> 4, pr = pl & 15;
      const int N = nt * 16 + pr;
#pragma unroll
      for (int s = 0; s < 16; s++) {
        const int K = kt * 32 + pg * 16 + s;
        wl[kt][nt][pl][s] = (__bf16)W[K * 128 + N];
      }
    }
    __syncthreads();

    for (int nt = 0; nt < 8; nt++) {
      v8f acc = {0.f, 0.f, 0.f, 0.f, 0.f, 0.f, 0.f, 0.f};
#pragma unroll
      for (int kt = 0; kt < 4; kt++) {
        v16bf bop = *(const v16bf*)(&wl[kt][nt][lane][0]);
        acc = wmma_bf16(a[kt].v, bop, acc);
      }
      const int chan = nt * 16 + lr;
      const int h = chan >> 6, d = chan & 63;
      const size_t trow = (((size_t)b * 2 + h) * 64 + d) * NTOK;
      if (m < 2) {
        __bf16* dst = (m == 0) ? qT : kT;
        float ss = 0.f;
#pragma unroll
        for (int j = 0; j < 8; j++) {
          const int tk = token0 + j + g * 8;
          const float val = acc[j];
          ss += val * val;
          dst[trow + tk] = (__bf16)val;
        }
        atomicAdd(((m == 0) ? ssq_q : ssq_k) + b * 128 + chan, ss);
      } else {
#pragma unroll
        for (int j = 0; j < 8; j++) {
          const int tk = token0 + j + g * 8;
          const float val = acc[j];
          vinpT[((size_t)b * 128 + chan) * NTOK + tk] = val;  // NCHW for convs
          const float gv = val * illu[((size_t)b * NTOK + tk) * 128 + chan];
          vT[trow + tk] = (__bf16)gv;
        }
      }
    }
  }
}

// ---------------- K2: attn_acc[b,h,d,e] += sum_n k[d,n] q[e,n] (split-K) ----
// grid 256 blocks (4 bh x 64 chunks of 1024 tokens), block 256 (8 waves)
__global__ __launch_bounds__(256) void k_attacc(const __bf16* __restrict__ kT,
                                                const __bf16* __restrict__ qT,
                                                float* __restrict__ acc) {
  const int bh = blockIdx.x >> 6;
  const int chunk = blockIdx.x & 63;
  const int wave = threadIdx.x >> 5, lane = threadIdx.x & 31;
  const int g = lane >> 4, lr = lane & 15;
  const int dt = wave & 3, half = wave >> 2;
  const int nbase = chunk * 1024 + half * 512;
  const __bf16* kbase = kT + (size_t)bh * 64 * NTOK;
  const __bf16* qbase = qT + (size_t)bh * 64 * NTOK;
  const int drow = dt * 16 + lr;

  v8f c[4];
#pragma unroll
  for (int et = 0; et < 4; et++) c[et] = (v8f){0.f, 0.f, 0.f, 0.f, 0.f, 0.f, 0.f, 0.f};

  for (int ks = 0; ks < 16; ks++) {
    const int n0 = nbase + ks * 32;
    V16 A;
    const __bf16* kp = kbase + (size_t)drow * NTOK + n0;
    A.h.lo = ld8(kp + g * 16);
    A.h.hi = ld8(kp + 8 + g * 16);
#pragma unroll
    for (int et = 0; et < 4; et++) {
      const int erow = et * 16 + lr;
      v16bf B = *(const v16bf*)(qbase + (size_t)erow * NTOK + n0 + g * 16);
      c[et] = wmma_bf16(A.v, B, c[et]);
    }
  }
  float* abase = acc + (size_t)bh * 64 * 64;
#pragma unroll
  for (int et = 0; et < 4; et++)
#pragma unroll
    for (int j = 0; j < 8; j++) {
      const int d = dt * 16 + j + g * 8;
      const int e = et * 16 + lr;
      atomicAdd(abase + d * 64 + e, c[et][j]);
    }
}

// ---------------- K3: normalize + rescale + softmax over e ----------------
__global__ __launch_bounds__(64) void k_softmax(const float* __restrict__ acc,
                                                const float* __restrict__ ssq_q,
                                                const float* __restrict__ ssq_k,
                                                const float* __restrict__ rescale,
                                                __bf16* __restrict__ attn_sm) {
  const int bh = blockIdx.x;
  const int b = bh >> 1, h = bh & 1;
  const int d = threadIdx.x;
  const float* row = acc + ((size_t)bh * 64 + d) * 64;
  const float invk = 1.f / fmaxf(sqrtf(ssq_k[b * 128 + h * 64 + d]), 1e-12f);
  const float rs = rescale[h];
  float vals[64];
  float mx = -INFINITY;
  for (int e = 0; e < 64; e++) {
    const float invq = 1.f / fmaxf(sqrtf(ssq_q[b * 128 + h * 64 + e]), 1e-12f);
    const float v = row[e] * invk * invq * rs;
    vals[e] = v;
    mx = fmaxf(mx, v);
  }
  float sum = 0.f;
  for (int e = 0; e < 64; e++) {
    vals[e] = expf(vals[e] - mx);
    sum += vals[e];
  }
  const float inv = 1.f / sum;
  __bf16* orow = attn_sm + ((size_t)bh * 64 + d) * 64;
  for (int e = 0; e < 64; e++) orow[e] = (__bf16)(vals[e] * inv);
}

// ---------------- K4: xo[d,n] = attn @ v ; store token-major bf16 ----------
// grid 256 blocks, block 256 (8 waves), each wave: 8 n-tiles
__global__ __launch_bounds__(256) void k_attnv(const __bf16* __restrict__ attn_sm,
                                               const __bf16* __restrict__ vT,
                                               __bf16* __restrict__ xoB) {
  const int w = blockIdx.x * 8 + (threadIdx.x >> 5);
  const int lane = threadIdx.x & 31, g = lane >> 4, lr = lane & 15;
  const int bh = w >> 9;
  const int widx = w & 511;
  const int b = bh >> 1, h = bh & 1;

  // preload A = attn rows (64x64, K split in 2)
  const __bf16* abase = attn_sm + (size_t)bh * 64 * 64;
  V16 A[4][2];
#pragma unroll
  for (int dt = 0; dt < 4; dt++)
#pragma unroll
    for (int kt = 0; kt < 2; kt++) {
      const int off = (dt * 16 + lr) * 64 + kt * 32;
      A[dt][kt].h.lo = ld8(abase + off + g * 16);
      A[dt][kt].h.hi = ld8(abase + off + 8 + g * 16);
    }

  const __bf16* vbase = vT + (size_t)bh * 64 * NTOK;
  for (int t = 0; t < 8; t++) {
    const int n0 = (widx * 8 + t) * 16;
    const int token = n0 + lr;
    v8f acc[4];
#pragma unroll
    for (int dt = 0; dt < 4; dt++) acc[dt] = (v8f){0.f, 0.f, 0.f, 0.f, 0.f, 0.f, 0.f, 0.f};
#pragma unroll
    for (int kt = 0; kt < 2; kt++) {
      v16bf B;
#pragma unroll
      for (int i = 0; i < 16; i++) {
        const int e = kt * 32 + g * 16 + i;
        B[i] = vbase[(size_t)e * NTOK + token];
      }
#pragma unroll
      for (int dt = 0; dt < 4; dt++) acc[dt] = wmma_bf16(A[dt][kt].v, B, acc[dt]);
    }
#pragma unroll
    for (int dt = 0; dt < 4; dt++)
#pragma unroll
      for (int j = 0; j < 8; j++) {
        const int d = dt * 16 + j + g * 8;
        xoB[((size_t)b * NTOK + token) * 128 + h * 64 + d] = (__bf16)acc[dt][j];
      }
  }
}

// ---------------- K5: out_c = xo @ Wp + bp  (writes '=') -------------------
__global__ __launch_bounds__(256) void k_proj(const __bf16* __restrict__ xoB,
                                              const float* __restrict__ Wp,
                                              const float* __restrict__ bp,
                                              float* __restrict__ out) {
  __shared__ __attribute__((aligned(32))) __bf16 wl[4][8][32][16];
  const int b = blockIdx.x >> 9;
  const int tbase = (blockIdx.x & 511) * 8;
  const int wave = threadIdx.x >> 5, lane = threadIdx.x & 31;
  const int g = lane >> 4, lr = lane & 15;

#pragma unroll
  for (int i = 0; i < 4; i++) {
    const int p = threadIdx.x + 256 * i;
    const int pl = p & 31, nt = (p >> 5) & 7, kt = p >> 8;
    const int pg = pl >> 4, pr = pl & 15;
    const int N = nt * 16 + pr;
#pragma unroll
    for (int s = 0; s < 16; s++) {
      const int K = kt * 32 + pg * 16 + s;
      wl[kt][nt][pl][s] = (__bf16)Wp[K * 128 + N];
    }
  }
  __syncthreads();

  const int token0 = (tbase + wave) * 16;
  const __bf16* xrow = xoB + ((size_t)b * NTOK + token0 + lr) * 128;
  V16 a[4];
#pragma unroll
  for (int kt = 0; kt < 4; kt++) {
    a[kt].h.lo = ld8(xrow + kt * 32 + g * 16);
    a[kt].h.hi = ld8(xrow + kt * 32 + 8 + g * 16);
  }
  for (int nt = 0; nt < 8; nt++) {
    v8f acc = {0.f, 0.f, 0.f, 0.f, 0.f, 0.f, 0.f, 0.f};
#pragma unroll
    for (int kt = 0; kt < 4; kt++) {
      v16bf bop = *(const v16bf*)(&wl[kt][nt][lane][0]);
      acc = wmma_bf16(a[kt].v, bop, acc);
    }
    const int chan = nt * 16 + lr;
    const float bias = bp[chan];
#pragma unroll
    for (int j = 0; j < 8; j++) {
      const int tk = token0 + j + g * 8;
      out[((size_t)b * NTOK + tk) * 128 + chan] = acc[j] + bias;
    }
  }
}

// ---------------- K6: fused dwconv3x3 -> exact GELU -> dwconv3x3, '+=' -----
__global__ __launch_bounds__(256) void k_conv(const float* __restrict__ vinpT,
                                              const float* __restrict__ w1,
                                              const float* __restrict__ w2,
                                              float* __restrict__ out) {
  const int idx = blockIdx.x;
  const int b = idx >> 13;
  const int chan = (idx >> 6) & 127;
  const int tile = idx & 63;
  const int y0 = (tile >> 3) * 32, x0 = (tile & 7) * 32;

  __shared__ float s_in[36][36];
  __shared__ float s_mid[34][34];

  const float* src = vinpT + ((size_t)b * 128 + chan) * NTOK;
  for (int i = threadIdx.x; i < 1296; i += 256) {
    const int yy = i / 36, xx = i % 36;
    const int y = y0 - 2 + yy, x = x0 - 2 + xx;
    s_in[yy][xx] =
        (y >= 0 && y < 256 && x >= 0 && x < 256) ? src[y * 256 + x] : 0.f;
  }
  float a1[9], a2[9];
#pragma unroll
  for (int t = 0; t < 9; t++) {
    a1[t] = w1[chan * 9 + t];
    a2[t] = w2[chan * 9 + t];
  }
  __syncthreads();
  for (int i = threadIdx.x; i < 1156; i += 256) {
    const int yy = i / 34, xx = i % 34;
    const int y = y0 - 1 + yy, x = x0 - 1 + xx;
    float s = 0.f;
    if (y >= 0 && y < 256 && x >= 0 && x < 256) {
#pragma unroll
      for (int dy = 0; dy < 3; dy++)
#pragma unroll
        for (int dx = 0; dx < 3; dx++) s += s_in[yy + dy][xx + dx] * a1[dy * 3 + dx];
      s = 0.5f * s * (1.f + erff(s * 0.70710678118654752f));  // exact GELU
    }
    s_mid[yy][xx] = s;
  }
  __syncthreads();
  for (int i = threadIdx.x; i < 1024; i += 256) {
    const int yy = i >> 5, xx = i & 31;
    float s = 0.f;
#pragma unroll
    for (int dy = 0; dy < 3; dy++)
#pragma unroll
      for (int dx = 0; dx < 3; dx++) s += s_mid[yy + dy][xx + dx] * a2[dy * 3 + dx];
    const int y = y0 + yy, x = x0 + xx;
    out[(((size_t)b * NTOK) + y * 256 + x) * 128 + chan] += s;
  }
}

// ---------------- launch ---------------------------------------------------
extern "C" void kernel_launch(void* const* d_in, const int* in_sizes, int n_in,
                              void* d_out, int out_size, void* d_ws,
                              size_t ws_size, hipStream_t stream) {
  const float* x = (const float*)d_in[0];
  const float* illu = (const float*)d_in[1];
  const float* Wq = (const float*)d_in[2];
  const float* Wk = (const float*)d_in[3];
  const float* Wv = (const float*)d_in[4];
  const float* rescale = (const float*)d_in[5];
  const float* Wp = (const float*)d_in[6];
  const float* bp = (const float*)d_in[7];
  const float* c1 = (const float*)d_in[8];
  const float* c2 = (const float*)d_in[9];
  float* out = (float*)d_out;

  char* ws = (char*)d_ws;
  constexpr size_t OFF_QT = 0;
  constexpr size_t OFF_KT = OFF_QT + (size_t)2 * 2 * 64 * NTOK * 2;    // bf16
  constexpr size_t OFF_VT = OFF_KT + (size_t)2 * 2 * 64 * NTOK * 2;    // bf16
  constexpr size_t OFF_XOB = OFF_VT + (size_t)2 * 2 * 64 * NTOK * 2;   // bf16
  constexpr size_t OFF_VINP = OFF_XOB + (size_t)2 * NTOK * 128 * 2;    // f32
  constexpr size_t OFF_ACC = OFF_VINP + (size_t)2 * 128 * NTOK * 4;    // f32
  constexpr size_t OFF_SSQ = OFF_ACC + (size_t)2 * 2 * 64 * 64 * 4;    // f32
  constexpr size_t OFF_ASM = OFF_SSQ + 2048;                           // bf16

  __bf16* qT = (__bf16*)(ws + OFF_QT);
  __bf16* kT = (__bf16*)(ws + OFF_KT);
  __bf16* vT = (__bf16*)(ws + OFF_VT);
  __bf16* xoB = (__bf16*)(ws + OFF_XOB);
  float* vinpT = (float*)(ws + OFF_VINP);
  float* attn_acc = (float*)(ws + OFF_ACC);
  float* ssq_q = (float*)(ws + OFF_SSQ);
  float* ssq_k = (float*)(ws + OFF_SSQ + 1024);
  __bf16* attn_sm = (__bf16*)(ws + OFF_ASM);

  // zero attn_acc (16384) + ssq_q (256) + ssq_k (256) = 16896 floats (contiguous)
  k_zero<<<66, 256, 0, stream>>>(attn_acc, 16896);
  k_qkv<<<1024, 256, 0, stream>>>(x, illu, Wq, Wk, Wv, qT, kT, vT, vinpT, ssq_q,
                                  ssq_k);
  k_attacc<<<256, 256, 0, stream>>>(kT, qT, attn_acc);
  k_softmax<<<4, 64, 0, stream>>>(attn_acc, ssq_q, ssq_k, rescale, attn_sm);
  k_attnv<<<256, 256, 0, stream>>>(attn_sm, vT, xoB);
  k_proj<<<1024, 256, 0, stream>>>(xoB, Wp, bp, out);
  k_conv<<<16384, 256, 0, stream>>>(vinpT, c1, c2, out);
}
